// iBlock_1537598292603
// MI455X (gfx1250) — matrix operand order repaired and checked by reading, not compiled, not run on hardware
//
#include <hip/hip_runtime.h>

// ---------------------------------------------------------------------------
// Problem constants (from the reference): B=16, PN=64, PL=128 -> BN=1024 rows,
// L=129 tokens, N_VAR=2, INNER=256, 4*INNER=1024, D_MODEL=256.
//
// Key algebraic collapse: only ctx[:,0] is consumed, the position-0 token is
// the all-zero emb_token, so Q[0]==0 exactly, softmax row 0 is uniform over
// masked positions, and the whole attention reduces to a masked mean of V,
// i.e. ((sum of masked 2-dim inputs) @ Wv) / count.  Exact, not approximate.
// ---------------------------------------------------------------------------

#define BN     1024   // B*PN sequences (rows of the activation matrix)
#define PL     128
#define LSEQ   129
#define INNER  256
#define HID    1024   // 4*INNER

typedef __attribute__((ext_vector_type(16))) __bf16 v16bf;
typedef __attribute__((ext_vector_type(8)))  __bf16 v8bf;
typedef __attribute__((ext_vector_type(8)))  float  v8f;

// ---------------------------------------------------------------------------
// Transpose + f32->bf16 convert:  W[R][C] (row-major f32) -> WT[C][R] (bf16)
// so GEMM B-fragments read contiguous K-runs.
// ---------------------------------------------------------------------------
__global__ void convert_transpose_kernel(const float* __restrict__ W,
                                         __bf16* __restrict__ WT,
                                         int R, int C)
{
    int idx = blockIdx.x * blockDim.x + threadIdx.x;
    if (idx >= R * C) return;
    int r = idx / C;
    int c = idx - r * C;
    WT[c * R + r] = (__bf16)W[idx];
}

// ---------------------------------------------------------------------------
// Per-sequence collapsed attention + LayerNorm.
// grid = BN, block = 128 (4 wave32).
//   last[b][j] = (S0*Wv[0][j] + S1*Wv[1][j]) / count      (f32, residual src)
//   h[b][j]    = LN(last[b])[j]*g[j]+beta[j]              (bf16, GEMM1 input)
// ---------------------------------------------------------------------------
__device__ inline float block_reduce128(float v, float* red)
{
    int t = threadIdx.x;
    red[t] = v;
    __syncthreads();
    #pragma unroll
    for (int s = 64; s > 0; s >>= 1) {
        if (t < s) red[t] += red[t + s];
        __syncthreads();
    }
    float r = red[0];
    __syncthreads();
    return r;
}

__global__ void attn_ln_kernel(const float* __restrict__ x,     // [BN][PL][2]
                               const int*   __restrict__ mask,  // [BN][LSEQ]
                               const float* __restrict__ Wv,    // [2][INNER]
                               const float* __restrict__ g,     // [INNER]
                               const float* __restrict__ beta,  // [INNER]
                               float*  __restrict__ last,       // [BN][INNER]
                               __bf16* __restrict__ hbf)        // [BN][INNER]
{
    __shared__ float red[128];
    const int b = blockIdx.x;
    const int p = threadIdx.x;             // 0..127 -> position p+1

    const int   mk  = mask[b * LSEQ + 1 + p];
    const float fx0 = x[(b * PL + p) * 2 + 0];
    const float fx1 = x[(b * PL + p) * 2 + 1];
    float s0  = mk ? fx0 : 0.f;
    float s1  = mk ? fx1 : 0.f;
    float cnt = (float)mk;
    if (p == 0) cnt += (float)mask[b * LSEQ];   // position 0 (value contrib = 0)

    const float S0 = block_reduce128(s0,  red);
    const float S1 = block_reduce128(s1,  red);
    const float C  = block_reduce128(cnt, red);
    const float ic = 1.f / C;

    // two columns per thread: j = p and j = p+128
    const float v0 = (S0 * Wv[p]       + S1 * Wv[INNER + p])       * ic;
    const float v1 = (S0 * Wv[p + 128] + S1 * Wv[INNER + p + 128]) * ic;
    last[b * INNER + p]       = v0;
    last[b * INNER + p + 128] = v1;

    const float sum = block_reduce128(v0 + v1, red);
    const float sq  = block_reduce128(v0 * v0 + v1 * v1, red);
    const float mu  = sum * (1.f / INNER);
    const float var = sq * (1.f / INNER) - mu * mu;
    const float rs  = rsqrtf(var + 1e-5f);

    hbf[b * INNER + p]       = (__bf16)((v0 - mu) * rs * g[p]       + beta[p]);
    hbf[b * INNER + p + 128] = (__bf16)((v1 - mu) * rs * g[p + 128] + beta[p + 128]);
}

// ---------------------------------------------------------------------------
// One 16x16 output tile accumulated over K with v_wmma_f32_16x16x32_bf16.
// A: [M][K] bf16 row-major.  BT: [N][K] bf16 (pre-transposed weight).
// Fragment layouts per CDNA5 ISA 7.12.2 (wave32):
//   A lane l: row = l&15; lanes 0-15 K {0..7,16..23}, lanes 16-31 K {8..15,24..31}
//   B lane l: col = l&15; lanes 0-15 K {0..7,16..23}, lanes 16-31 K {8..15,24..31}
//   C lane l: col = l&15; VGPR i -> row i (lanes 0-15) / row i+8 (lanes 16-31)
// ---------------------------------------------------------------------------
__device__ inline v8f wmma_tile(const __bf16* __restrict__ A,
                                const __bf16* __restrict__ BT,
                                int K, int row0, int col0, int lane)
{
    const int r  = row0 + (lane & 15);
    const int c  = col0 + (lane & 15);
    const int ah = (lane >> 4) << 3;      // A K-offset within step: 0 or 8
    const int bh = (lane >> 4) << 3;      // B K-offset within step: 0 or 8
    const __bf16* pa = A  + (long)r * K + ah;
    const __bf16* pb = BT + (long)c * K + bh;

    v8f acc = {};
    for (int kk = 0; kk < K; kk += 32) {
        v8bf a_lo = *reinterpret_cast<const v8bf*>(pa + kk);        // K+0..7
        v8bf a_hi = *reinterpret_cast<const v8bf*>(pa + kk + 16);   // K+16..23
        v8bf b_lo = *reinterpret_cast<const v8bf*>(pb + kk);        // K+0..7
        v8bf b_hi = *reinterpret_cast<const v8bf*>(pb + kk + 16);   // K+16..23
        v16bf a = __builtin_shufflevector(a_lo, a_hi, 0,1,2,3,4,5,6,7,
                                                      8,9,10,11,12,13,14,15);
        v16bf bm = __builtin_shufflevector(b_lo, b_hi, 0,1,2,3,4,5,6,7,
                                                       8,9,10,11,12,13,14,15);
        acc = __builtin_amdgcn_wmma_f32_16x16x32_bf16(
                  false, a, false, bm, (short)0, acc, false, false);
    }
    return acc;
}

// GEMM1: A1 = relu(h @ W1 + b1)   (M=1024, K=256, N=1024) -> bf16
__global__ __launch_bounds__(128)
void gemm_relu_kernel(const __bf16* __restrict__ H,
                      const __bf16* __restrict__ W1T,
                      const float*  __restrict__ b1,
                      __bf16* __restrict__ A1,
                      int M, int N, int K)
{
    const int lane = threadIdx.x & 31;
    const int tile = blockIdx.x * (blockDim.x >> 5) + (threadIdx.x >> 5);
    const int tn_n = N >> 4;
    const int tm = tile / tn_n, tn = tile - tm * tn_n;

    v8f acc = wmma_tile(H, W1T, K, tm * 16, tn * 16, lane);

    const int col  = tn * 16 + (lane & 15);
    const int row0 = tm * 16 + ((lane >> 4) << 3);
    const float bias = b1[col];
    #pragma unroll
    for (int i = 0; i < 8; ++i) {
        float v = acc[i] + bias;
        v = v > 0.f ? v : 0.f;
        A1[(long)(row0 + i) * N + col] = (__bf16)v;
    }
}

// GEMM2: last2 = last + (A1 @ W2 + b2)   (M=1024, K=1024, N=256) -> bf16
__global__ __launch_bounds__(128)
void gemm_residual_kernel(const __bf16* __restrict__ A1,
                          const __bf16* __restrict__ W2T,
                          const float*  __restrict__ b2,
                          const float*  __restrict__ last,
                          __bf16* __restrict__ L2,
                          int M, int N, int K)
{
    const int lane = threadIdx.x & 31;
    const int tile = blockIdx.x * (blockDim.x >> 5) + (threadIdx.x >> 5);
    const int tn_n = N >> 4;
    const int tm = tile / tn_n, tn = tile - tm * tn_n;

    v8f acc = wmma_tile(A1, W2T, K, tm * 16, tn * 16, lane);

    const int col  = tn * 16 + (lane & 15);
    const int row0 = tm * 16 + ((lane >> 4) << 3);
    const float bias = b2[col];
    #pragma unroll
    for (int i = 0; i < 8; ++i) {
        float v = acc[i] + bias + last[(long)(row0 + i) * N + col];
        L2[(long)(row0 + i) * N + col] = (__bf16)v;
    }
}

// GEMM3: out = last2 @ Wo   (M=1024, K=256, N=256) -> f32
__global__ __launch_bounds__(128)
void gemm_out_kernel(const __bf16* __restrict__ L2,
                     const __bf16* __restrict__ WoT,
                     float* __restrict__ out,
                     int M, int N, int K)
{
    const int lane = threadIdx.x & 31;
    const int tile = blockIdx.x * (blockDim.x >> 5) + (threadIdx.x >> 5);
    const int tn_n = N >> 4;
    const int tm = tile / tn_n, tn = tile - tm * tn_n;

    v8f acc = wmma_tile(L2, WoT, K, tm * 16, tn * 16, lane);

    const int col  = tn * 16 + (lane & 15);
    const int row0 = tm * 16 + ((lane >> 4) << 3);
    #pragma unroll
    for (int i = 0; i < 8; ++i)
        out[(long)(row0 + i) * N + col] = acc[i];
}

// ---------------------------------------------------------------------------
// Host launcher.  Workspace layout (bytes, all 256-aligned), total 5,373,952 (5.125 MiB):
//   last  f32  [1024][256]   : 1,048,576
//   h     bf16 [1024][256]   :   524,288
//   W1T   bf16 [1024][256]   :   524,288
//   W2T   bf16 [256][1024]   :   524,288
//   WoT   bf16 [256][256]    :   131,072
//   A1    bf16 [1024][1024]  : 2,097,152
//   L2    bf16 [1024][256]   :   524,288
// ---------------------------------------------------------------------------
extern "C" void kernel_launch(void* const* d_in, const int* in_sizes, int n_in,
                              void* d_out, int out_size, void* d_ws, size_t ws_size,
                              hipStream_t stream)
{
    const float* x    = (const float*)d_in[0];
    const int*   mask = (const int*)  d_in[2];
    const float* Wv   = (const float*)d_in[6];
    const float* Wo   = (const float*)d_in[7];
    const float* ln_g = (const float*)d_in[8];
    const float* ln_b = (const float*)d_in[9];
    const float* W1   = (const float*)d_in[10];
    const float* b1   = (const float*)d_in[11];
    const float* W2   = (const float*)d_in[12];
    const float* b2   = (const float*)d_in[13];
    float* out = (float*)d_out;

    char* ws = (char*)d_ws;
    size_t off = 0;
    float*  last = (float*) (ws + off); off += (size_t)BN * INNER * 4;
    __bf16* h    = (__bf16*)(ws + off); off += (size_t)BN * INNER * 2;
    __bf16* W1T  = (__bf16*)(ws + off); off += (size_t)INNER * HID * 2;
    __bf16* W2T  = (__bf16*)(ws + off); off += (size_t)HID * INNER * 2;
    __bf16* WoT  = (__bf16*)(ws + off); off += (size_t)INNER * INNER * 2;
    __bf16* A1   = (__bf16*)(ws + off); off += (size_t)BN * HID * 2;
    __bf16* L2   = (__bf16*)(ws + off); off += (size_t)BN * INNER * 2;
    (void)ws_size; (void)in_sizes; (void)n_in; (void)out_size;

    // 1) weight transpose + bf16 convert
    convert_transpose_kernel<<<(INNER * HID + 255) / 256, 256, 0, stream>>>(W1, W1T, INNER, HID);
    convert_transpose_kernel<<<(HID * INNER + 255) / 256, 256, 0, stream>>>(W2, W2T, HID, INNER);
    convert_transpose_kernel<<<(INNER * INNER + 255) / 256, 256, 0, stream>>>(Wo, WoT, INNER, INNER);

    // 2) collapsed attention + LayerNorm
    attn_ln_kernel<<<BN, 128, 0, stream>>>(x, mask, Wv, ln_g, ln_b, last, h);

    // 3) MLP up-projection + ReLU   (64x64 tiles, 4 waves/block)
    gemm_relu_kernel<<<(64 * 64) / 4, 128, 0, stream>>>(h, W1T, b1, A1, BN, HID, INNER);

    // 4) MLP down-projection + bias + residual   (64x16 tiles)
    gemm_residual_kernel<<<(64 * 16) / 4, 128, 0, stream>>>(A1, W2T, b2, last, L2, BN, INNER, HID);

    // 5) output projection   (64x16 tiles)
    gemm_out_kernel<<<(64 * 16) / 4, 128, 0, stream>>>(L2, WoT, out, BN, INNER, INNER);
}
